// DeepNearestClassMean_39298950758648
// MI455X (gfx1250) — compile-verified, matrix operand order checked by reading, not run on hardware
//
#include <hip/hip_runtime.h>

// ---------------------------------------------------------------------------
// DeepNearestClassMean on MI455X (gfx1250, wave32)
// out[b,c] = -(||x_b||^2 + ||m_c||^2 - 2 x.mT) = 2*cross - xsq - msq
// GEMM via split-precision bf16 WMMA (hi/lo decomposition, 3 WMMAs / K32 chunk)
// ---------------------------------------------------------------------------

typedef __attribute__((ext_vector_type(16))) __bf16 v16bf;
typedef __attribute__((ext_vector_type(8)))  __bf16 v8bf;
typedef __attribute__((ext_vector_type(8)))  float  v8f;

#define TILE_M     128
#define TILE_N     128
#define TILE_K     32
#define LDS_STRIDE 40   // 32 K elems + 8 bf16 pad -> conflict-free b128 reads

// Load 16 consecutive floats (64B) from global, scaled.
__device__ __forceinline__ void ld16(const float* __restrict__ g, float s,
                                     float* __restrict__ f) {
    const float4* p4 = (const float4*)g;
#pragma unroll
    for (int i = 0; i < 4; ++i) {
        float4 v = p4[i];
        f[4 * i + 0] = s * v.x;
        f[4 * i + 1] = s * v.y;
        f[4 * i + 2] = s * v.z;
        f[4 * i + 3] = s * v.w;
    }
}

// Split 16 fp32 into bf16 hi + bf16 lo and store 2x16B each to LDS.
__device__ __forceinline__ void split_store(const float* __restrict__ f,
                                            __bf16* __restrict__ hiP,
                                            __bf16* __restrict__ loP) {
    v8bf h0{}, h1{}, l0{}, l1{};
#pragma unroll
    for (int i = 0; i < 8; ++i) {
        __bf16 hh = (__bf16)f[i];
        h0[i] = hh;
        l0[i] = (__bf16)(f[i] - (float)hh);
    }
#pragma unroll
    for (int i = 0; i < 8; ++i) {
        __bf16 hh = (__bf16)f[8 + i];
        h1[i] = hh;
        l1[i] = (__bf16)(f[8 + i] - (float)hh);
    }
    *(v8bf*)(hiP)     = h0;
    *(v8bf*)(hiP + 8) = h1;
    *(v8bf*)(loP)     = l0;
    *(v8bf*)(loP + 8) = l1;
}

// WMMA 16-bit A/B fragment from a row-major [rows][LDS_STRIDE] LDS tile.
// ISA layout: lanes 0-15 -> row = lane,    K = {0..7, 16..23}
//             lanes16-31 -> row = lane-16, K = {8..15, 24..31}
__device__ __forceinline__ v16bf load_frag(const __bf16* __restrict__ tile,
                                           int row, int lane) {
    int kh = (lane & 16) ? 8 : 0;
    const __bf16* p = tile + row * LDS_STRIDE + kh;
    v8bf c0 = *(const v8bf*)(p);        // K chunk +0
    v8bf c1 = *(const v8bf*)(p + 16);   // K chunk +16
    return __builtin_shufflevector(c0, c1, 0, 1, 2, 3, 4, 5, 6, 7,
                                           8, 9, 10, 11, 12, 13, 14, 15);
}

// ---------------------------------------------------------------------------
// Row squared-norm: one block per row.
// ---------------------------------------------------------------------------
__global__ __launch_bounds__(256) void dncm_rownorm(const float* __restrict__ a,
                                                    float* __restrict__ out,
                                                    int cols) {
    const float4* p = (const float4*)(a + (size_t)blockIdx.x * cols);
    float s = 0.0f;
    for (int i = threadIdx.x; i < (cols >> 2); i += blockDim.x) {
        float4 v = p[i];
        s += v.x * v.x + v.y * v.y + v.z * v.z + v.w * v.w;
    }
    __shared__ float red[256];
    red[threadIdx.x] = s;
    __syncthreads();
    for (int off = 128; off > 0; off >>= 1) {
        if ((int)threadIdx.x < off) red[threadIdx.x] += red[threadIdx.x + off];
        __syncthreads();
    }
    if (threadIdx.x == 0) out[blockIdx.x] = red[0];
}

// ---------------------------------------------------------------------------
// Main split-bf16 WMMA GEMM + distance epilogue.
// Block: 256 threads = 8 waves, 128x128 output tile.
// Wave (wm,wn): wm in 0..3 (32 rows), wn in 0..1 (64 cols) -> 2x4 16x16 tiles.
// ---------------------------------------------------------------------------
__global__ __launch_bounds__(256)
void dncm_gemm(const float* __restrict__ x, const float* __restrict__ means,
               const float* __restrict__ xsq, const float* __restrict__ msq,
               float* __restrict__ out, int B, int C, int F) {
    __shared__ alignas(16) __bf16 Ahi[TILE_M * LDS_STRIDE];
    __shared__ alignas(16) __bf16 Alo[TILE_M * LDS_STRIDE];
    __shared__ alignas(16) __bf16 Bhi[TILE_N * LDS_STRIDE];
    __shared__ alignas(16) __bf16 Blo[TILE_N * LDS_STRIDE];

    const int t    = threadIdx.x;
    const int lane = t & 31;
    const int wave = t >> 5;
    const int wm   = wave >> 1;  // 0..3
    const int wn   = wave & 1;   // 0..1
    const int bN   = blockIdx.x;
    const int bM   = blockIdx.y;

    v8f acc[2][4];
#pragma unroll
    for (int m = 0; m < 2; ++m)
#pragma unroll
        for (int n = 0; n < 4; ++n) acc[m][n] = v8f{};

    // Staging assignment: thread -> (row r, K-half kh) of the 128x32 tiles.
    const int r  = t >> 1;
    const int kh = (t & 1) << 4;  // 0 or 16

    int arowi = bM * TILE_M + r;
    if (arowi >= B) arowi = B - 1;      // B=8192 divides; defensive clamp
    int crowi = bN * TILE_N + r;
    if (crowi >= C) crowi = C - 1;      // C=10000 is not a tile multiple

    const float* xpBase = x + (size_t)arowi * F + kh;
    const float* mpBase = means + (size_t)crowi * F + kh;
    __bf16* aH = &Ahi[r * LDS_STRIDE + kh];
    __bf16* aL = &Alo[r * LDS_STRIDE + kh];
    __bf16* bH = &Bhi[r * LDS_STRIDE + kh];
    __bf16* bL = &Blo[r * LDS_STRIDE + kh];

    for (int k0 = 0; k0 < F; k0 += TILE_K) {
        float fx[16], fm[16];
        ld16(xpBase + k0, 2.0f, fx);    // fold the "2*cross" scale into A (exact)
        ld16(mpBase + k0, 1.0f, fm);

        __syncthreads();                // prior chunk's LDS reads complete
        split_store(fx, aH, aL);
        split_store(fm, bH, bL);
        __syncthreads();                // tile staged

        v16bf aHf[2], aLf[2];
#pragma unroll
        for (int m = 0; m < 2; ++m) {
            int row = wm * 32 + m * 16 + (lane & 15);
            aHf[m] = load_frag(Ahi, row, lane);
            aLf[m] = load_frag(Alo, row, lane);
        }
#pragma unroll
        for (int n = 0; n < 4; ++n) {
            int col = wn * 64 + n * 16 + (lane & 15);
            v16bf bHf = load_frag(Bhi, col, lane);
            v16bf bLf = load_frag(Blo, col, lane);
#pragma unroll
            for (int m = 0; m < 2; ++m) {
                // cross = hi*hi + hi*lo + lo*hi  (drop lo*lo ~ 2^-16 rel.)
                acc[m][n] = __builtin_amdgcn_wmma_f32_16x16x32_bf16(
                    false, aHf[m], false, bHf, (short)0, acc[m][n], false, false);
                acc[m][n] = __builtin_amdgcn_wmma_f32_16x16x32_bf16(
                    false, aHf[m], false, bLf, (short)0, acc[m][n], false, false);
                acc[m][n] = __builtin_amdgcn_wmma_f32_16x16x32_bf16(
                    false, aLf[m], false, bHf, (short)0, acc[m][n], false, false);
            }
        }
    }

    // Epilogue: out[b,c] = acc(=2*cross) - xsq[b] - msq[c]
    // C/D layout: VGPR j, lanes 0-15 -> M=j, lanes 16-31 -> M=j+8; N = lane&15.
    const size_t rowBase = (size_t)bM * TILE_M + wm * 32 + ((lane >> 4) << 3);
    const int    colBase = bN * TILE_N + wn * 64 + (lane & 15);
#pragma unroll
    for (int n = 0; n < 4; ++n) {
        int col = colBase + n * 16;
        if (col < C) {
            float msqv = msq[col];
#pragma unroll
            for (int m = 0; m < 2; ++m) {
#pragma unroll
                for (int j = 0; j < 8; ++j) {
                    size_t row = rowBase + m * 16 + j;
                    out[row * (size_t)C + col] = acc[m][n][j] - xsq[row] - msqv;
                }
            }
        }
    }
}

// ---------------------------------------------------------------------------
extern "C" void kernel_launch(void* const* d_in, const int* in_sizes, int n_in,
                              void* d_out, int out_size, void* d_ws, size_t ws_size,
                              hipStream_t stream) {
    const float* x     = (const float*)d_in[0];
    const float* means = (const float*)d_in[1];
    float*       out   = (float*)d_out;

    const int F = 2048;
    const int B = in_sizes[0] / F;   // 8192
    const int C = in_sizes[1] / F;   // 10000

    float* xsq = (float*)d_ws;       // [B]
    float* msq = xsq + B;            // [C]  (B+C floats = ~73 KB scratch)

    dncm_rownorm<<<B, 256, 0, stream>>>(x, xsq, F);
    dncm_rownorm<<<C, 256, 0, stream>>>(means, msq, F);

    dim3 grid((C + TILE_N - 1) / TILE_N, (B + TILE_M - 1) / TILE_M);
    dncm_gemm<<<grid, 256, 0, stream>>>(x, means, xsq, msq, out, B, C, F);
}
